// DendriteInput_19602230739124
// MI455X (gfx1250) — compile-verified, matrix operand order checked by reading, not run on hardware
//
#include <hip/hip_runtime.h>
#include <hip/hip_bf16.h>

typedef __attribute__((ext_vector_type(16))) __bf16 v16bf;
typedef __attribute__((ext_vector_type(8)))  float  v8f;

#define IN_DIM      2048
#define N_DEND      8192
#define BATCH       8192
#define TOPK        819
#define BOOST_STR   2.0f
#define PCT_ON      0.1f

// ---------- helpers ----------
__device__ __forceinline__ unsigned short f32_to_bf16_bits(float f) {
    union { float f; unsigned u; } c; c.f = f;
    unsigned u = c.u;
    unsigned r = u + 0x7FFFu + ((u >> 16) & 1u);   // round-to-nearest-even
    return (unsigned short)(r >> 16);
}
__device__ __forceinline__ float bf16_bits_to_f32(unsigned short h) {
    union { unsigned u; float f; } c; c.u = ((unsigned)h) << 16;
    return c.f;
}
// monotone float -> uint key (order preserving)
__device__ __forceinline__ unsigned key_of(float f) {
    union { float f; unsigned u; } c; c.f = f;
    unsigned u = c.u;
    return (u & 0x80000000u) ? ~u : (u | 0x80000000u);
}

// split one float4 into hi/lo bf16 quads and store to LDS (8-byte stores)
__device__ __forceinline__ void split_store4(unsigned short* hi, unsigned short* lo, float4 v) {
    ushort4 h, l;
    h.x = f32_to_bf16_bits(v.x); l.x = f32_to_bf16_bits(v.x - bf16_bits_to_f32(h.x));
    h.y = f32_to_bf16_bits(v.y); l.y = f32_to_bf16_bits(v.y - bf16_bits_to_f32(h.y));
    h.z = f32_to_bf16_bits(v.z); l.z = f32_to_bf16_bits(v.z - bf16_bits_to_f32(h.z));
    h.w = f32_to_bf16_bits(v.w); l.w = f32_to_bf16_bits(v.w - bf16_bits_to_f32(h.w));
    *(ushort4*)hi = h;
    *(ushort4*)lo = l;
}

// Load a 16x32 bf16 fragment from a [128][32] ushort LDS tile.
// Lane l (wave32): row = base + (l&15); K-half = l>>4.
// elements 0..7  -> K = kh*8 + 0..7      (16 contiguous bytes)
// elements 8..15 -> K = kh*8 + 16..23    (16 contiguous bytes)
__device__ __forceinline__ v16bf load_frag(const unsigned short* tile, int row, int kh) {
    union { uint4 q[2]; v16bf v; } u;
    const unsigned short* p = tile + row * 32 + kh * 8;
    u.q[0] = *(const uint4*)(p);
    u.q[1] = *(const uint4*)(p + 16);
    return u.v;
}

// ---------- kernel 0: boost factors ----------
__global__ __launch_bounds__(256) void boost_kernel(const float* __restrict__ duty,
                                                    float* __restrict__ boost) {
    int n = blockIdx.x * 256 + threadIdx.x;
    if (n < N_DEND) boost[n] = __expf(BOOST_STR * (PCT_ON - duty[n]));
}

// ---------- kernel 1: masked GEMM via bf16x3-split WMMA ----------
// y[B][N] = x[B][D] . (W*mask)[N][D]^T + bias[N]
// block tile 128(M) x 128(N), 8 waves (2x4), wave tile 64x32, K-step 32.
__global__ __launch_bounds__(256) void gemm_wmma_bf16x3(
        const float* __restrict__ x, const float* __restrict__ w,
        const unsigned char* __restrict__ mask, const float* __restrict__ bias,
        float* __restrict__ y) {
    __shared__ __align__(16) unsigned short sAhi[128 * 32];
    __shared__ __align__(16) unsigned short sAlo[128 * 32];
    __shared__ __align__(16) unsigned short sBhi[128 * 32];
    __shared__ __align__(16) unsigned short sBlo[128 * 32];

    const int tid  = threadIdx.x;
    const int lane = tid & 31;
    const int wid  = tid >> 5;
    const int wm   = wid >> 2;     // 0..1  (M direction, 64 rows each)
    const int wn   = wid & 3;      // 0..3  (N direction, 32 cols each)
    const int m0   = blockIdx.x * 128;
    const int n0   = blockIdx.y * 128;
    const int lm   = lane & 15;
    const int kh   = lane >> 4;

    v8f acc[4][2];
    for (int i = 0; i < 4; ++i)
        for (int j = 0; j < 2; ++j)
            acc[i][j] = (v8f){0.f, 0.f, 0.f, 0.f, 0.f, 0.f, 0.f, 0.f};

    for (int kc = 0; kc < IN_DIM; kc += 32) {
        // cooperative tile load: 128x32 f32 for A and masked B; split to hi/lo bf16
        #pragma unroll
        for (int it = 0; it < 4; ++it) {
            int f4 = tid + it * 256;        // 0..1023 float4 slots
            int r  = f4 >> 3;               // 0..127
            int c4 = f4 & 7;                // 0..7 (float4 column)
            int o  = r * 32 + c4 * 4;

            size_t ga = (size_t)(m0 + r) * IN_DIM + kc + c4 * 4;
            float4 xv = *(const float4*)(x + ga);
            split_store4(sAhi + o, sAlo + o, xv);

            size_t gb = (size_t)(n0 + r) * IN_DIM + kc + c4 * 4;
            float4 wv = *(const float4*)(w + gb);
            unsigned mv = *(const unsigned*)(mask + gb);   // 4 bool bytes
            wv.x = (mv & 0x000000FFu) ? wv.x : 0.0f;
            wv.y = (mv & 0x0000FF00u) ? wv.y : 0.0f;
            wv.z = (mv & 0x00FF0000u) ? wv.z : 0.0f;
            wv.w = (mv & 0xFF000000u) ? wv.w : 0.0f;
            split_store4(sBhi + o, sBlo + o, wv);

            // hint the next K-chunk toward the caches
            if (kc + 32 < IN_DIM) {
                __builtin_prefetch(x + ga + 32, 0, 1);
                __builtin_prefetch(w + gb + 32, 0, 1);
            }
        }
        __syncthreads();

        // B fragments for this wave's two N-subtiles
        v16bf bh[2], bl[2];
        #pragma unroll
        for (int j = 0; j < 2; ++j) {
            int row = wn * 32 + j * 16 + lm;
            bh[j] = load_frag(sBhi, row, kh);
            bl[j] = load_frag(sBlo, row, kh);
        }
        #pragma unroll
        for (int i = 0; i < 4; ++i) {
            int row = wm * 64 + i * 16 + lm;
            v16bf ah = load_frag(sAhi, row, kh);
            v16bf al = load_frag(sAlo, row, kh);
            #pragma unroll
            for (int j = 0; j < 2; ++j) {
                acc[i][j] = __builtin_amdgcn_wmma_f32_16x16x32_bf16(
                    false, ah, false, bh[j], (short)0, acc[i][j], false, false);
                acc[i][j] = __builtin_amdgcn_wmma_f32_16x16x32_bf16(
                    false, ah, false, bl[j], (short)0, acc[i][j], false, false);
                acc[i][j] = __builtin_amdgcn_wmma_f32_16x16x32_bf16(
                    false, al, false, bh[j], (short)0, acc[i][j], false, false);
            }
        }
        __syncthreads();
    }

    // epilogue: + bias, store f32.  C/D layout: vgpr r -> M = r + (lane>>4)*8, N = lane&15
    #pragma unroll
    for (int i = 0; i < 4; ++i) {
        #pragma unroll
        for (int j = 0; j < 2; ++j) {
            int n  = n0 + wn * 32 + j * 16 + lm;
            float bv = bias[n];
            int mr = m0 + wm * 64 + i * 16 + kh * 8;
            #pragma unroll
            for (int r = 0; r < 8; ++r)
                y[(size_t)(mr + r) * N_DEND + n] = acc[i][j][r] + bv;
        }
    }
}

// ---------- kernel 2: per-row K-th largest boosted key (radix select) ----------
__global__ __launch_bounds__(256) void select_kernel(const float* __restrict__ y,
                                                     const float* __restrict__ boost,
                                                     unsigned* __restrict__ thr) {
    __shared__ unsigned hist[256];
    __shared__ unsigned s_prefix;
    __shared__ unsigned s_remaining;
    const int tid = threadIdx.x;
    const float* row = y + (size_t)blockIdx.x * N_DEND;
    if (tid == 0) { s_prefix = 0u; s_remaining = TOPK; }

    #pragma unroll
    for (int shift = 24; shift >= 0; shift -= 8) {
        hist[tid] = 0u;
        __syncthreads();
        unsigned prefix = s_prefix;
        unsigned hmask  = (shift == 24) ? 0u : (0xFFFFFFFFu << (shift + 8));
        for (int n = tid; n < N_DEND; n += 256) {
            unsigned k = key_of(row[n] * boost[n]);
            if (((k ^ prefix) & hmask) == 0u)
                atomicAdd(&hist[(k >> shift) & 255u], 1u);
        }
        __syncthreads();
        if (tid == 0) {
            unsigned c = 0, rem = s_remaining; int sel = 0;
            for (int bin = 255; bin >= 0; --bin) {
                unsigned h = hist[bin];
                if (c + h >= rem) { sel = bin; s_remaining = rem - c; break; }
                c += h;
            }
            s_prefix = prefix | ((unsigned)sel << shift);
        }
        __syncthreads();
    }
    if (tid == 0) thr[blockIdx.x] = s_prefix;
}

// ---------- kernel 3: apply winner mask in place ----------
__global__ __launch_bounds__(256) void apply_kernel(float* __restrict__ y,
                                                    const float* __restrict__ boost,
                                                    const unsigned* __restrict__ thr) {
    size_t idx = (size_t)blockIdx.x * 256 + threadIdx.x;
    int b = (int)(idx >> 13);        // /8192
    int n = (int)(idx & 8191);
    float v = y[idx];
    unsigned k = key_of(v * boost[n]);
    y[idx] = (k >= thr[b]) ? v : 0.0f;
}

extern "C" void kernel_launch(void* const* d_in, const int* in_sizes, int n_in,
                              void* d_out, int out_size, void* d_ws, size_t ws_size,
                              hipStream_t stream) {
    const float*         x    = (const float*)d_in[0];          // [8192,2048]
    const float*         w    = (const float*)d_in[1];          // [8192,2048]
    const float*         bias = (const float*)d_in[2];          // [8192]
    const float*         duty = (const float*)d_in[3];          // [8192]
    const unsigned char* mask = (const unsigned char*)d_in[4];  // [8192,2048] bool bytes
    float*    y     = (float*)d_out;                            // [8192,8192]
    float*    boost = (float*)d_ws;                             // 8192 floats
    unsigned* thr   = (unsigned*)d_ws + N_DEND;                 // 8192 keys

    boost_kernel<<<N_DEND / 256, 256, 0, stream>>>(duty, boost);
    gemm_wmma_bf16x3<<<dim3(BATCH / 128, N_DEND / 128), 256, 0, stream>>>(x, w, mask, bias, y);
    select_kernel<<<BATCH, 256, 0, stream>>>(y, boost, thr);
    apply_kernel<<<(unsigned)(((size_t)BATCH * N_DEND) / 256), 256, 0, stream>>>(y, boost, thr);
}
